// LocalWindowAttention_62835371541012
// MI455X (gfx1250) — matrix operand order repaired and checked by reading, not compiled, not run on hardware
//
#include <hip/hip_runtime.h>

#define DIM 256
#define WS  64
#define NH  8
#define HD  32
#define QKSCALE 0.17677669529663687f  // 32^-0.5

typedef __attribute__((ext_vector_type(4)))  float          v4f;
typedef __attribute__((ext_vector_type(8)))  float          v8f;
typedef __attribute__((ext_vector_type(8)))  unsigned short v8u;
typedef __attribute__((ext_vector_type(16))) unsigned short v16u;
typedef __attribute__((ext_vector_type(16))) __bf16         v16bf;

#define SXS (DIM + 8)        // 264: X tile / attn-out tile row stride (bf16)
#define SQS (2 * DIM + 8)    // 520: [Q|K] row stride
#define SVS (WS + 8)         // 72 : V-transposed row stride
#define SPS (WS + 8)         // 72 : per-wave softmax-prob row stride

#define WQKV_ELEMS (3 * DIM * DIM)   // 196608
#define WPROJ_ELEMS (DIM * DIM)      // 65536
#define WS_BYTES_NEEDED ((size_t)(WQKV_ELEMS + WPROJ_ELEMS) * 2)

// Native bf16 convert (RNE fptrunc) -> hardware cvt on gfx1250.
__device__ __forceinline__ unsigned short f2bf(float f) {
  __bf16 b = (__bf16)f;
  return __builtin_bit_cast(unsigned short, b);
}

// A-matrix fragment (16x32 bf16): lane m = lane&15; lanes 0-15 hold K=0..15,
// lanes 16-31 hold K=16..31 -> one contiguous 16-element run per lane.
__device__ __forceinline__ v16bf frag_a(const unsigned short* row, int cb, int hf) {
  v8u lo = *(const v8u*)(row + cb + hf * 16);
  v8u hi = *(const v8u*)(row + cb + hf * 16 + 8);
  v16u u = __builtin_shufflevector(lo, hi, 0,1,2,3,4,5,6,7,8,9,10,11,12,13,14,15);
  return __builtin_bit_cast(v16bf, u);
}

// B-matrix fragment (32x16 bf16): lane n = lane&15; lanes 0-15 hold K=0..15,
// lanes 16-31 hold K=16..31 -> one contiguous 16-element run per lane.
__device__ __forceinline__ v16bf frag_b_bf(const unsigned short* row, int cb, int hf) {
  v8u lo = *(const v8u*)(row + cb + hf * 16);
  v8u hi = *(const v8u*)(row + cb + hf * 16 + 8);
  v16u u = __builtin_shufflevector(lo, hi, 0,1,2,3,4,5,6,7,8,9,10,11,12,13,14,15);
  return __builtin_bit_cast(v16bf, u);
}

// Fallback: B fragment sourced from f32 weights, converted in registers.
__device__ __forceinline__ v16bf frag_b_w(const float* row, int cb, int hf) {
  const v4f* p = (const v4f*)(row + cb + hf * 16);
  v4f a = p[0], b = p[1], c = p[2], d = p[3];
  v16u u = (v16u)0;
#pragma unroll
  for (int e = 0; e < 4; ++e) {
    u[e]      = f2bf(a[e]);
    u[e + 4]  = f2bf(b[e]);
    u[e + 8]  = f2bf(c[e]);
    u[e + 12] = f2bf(d[e]);
  }
  return __builtin_bit_cast(v16bf, u);
}

__device__ __forceinline__ v8f wmma_bf16(v16bf a, v16bf b, v8f c) {
  return __builtin_amdgcn_wmma_f32_16x16x32_bf16(false, a, false, b, (short)0, c,
                                                 false, false);
}

// -------- prep: one-shot f32 -> bf16 weight conversion into d_ws --------
__global__ void conv_weights_kernel(const float* __restrict__ w_qkv,
                                    const float* __restrict__ w_proj,
                                    unsigned short* __restrict__ dst) {
  int i = blockIdx.x * blockDim.x + threadIdx.x;   // 65536 threads, 4 elems each
  v4f v;
  unsigned short* o;
  if (i < WQKV_ELEMS / 4) {
    v = ((const v4f*)w_qkv)[i];
    o = dst + i * 4;
  } else {
    int j = i - WQKV_ELEMS / 4;
    v = ((const v4f*)w_proj)[j];
    o = dst + WQKV_ELEMS + j * 4;
  }
  unsigned* p = (unsigned*)o;
  p[0] = (unsigned)f2bf(v[0]) | ((unsigned)f2bf(v[1]) << 16);
  p[1] = (unsigned)f2bf(v[2]) | ((unsigned)f2bf(v[3]) << 16);
}

template <bool WBF>
__global__ __launch_bounds__(256, 1) void lwa_fused_kernel(
    const float* __restrict__ x, const float* __restrict__ w_qkv,
    const float* __restrict__ w_proj, const float* __restrict__ b_proj,
    const float* __restrict__ rel_bias,
    const unsigned short* __restrict__ wqkv_bf,
    const unsigned short* __restrict__ wproj_bf,
    float* __restrict__ out) {
  __shared__ __align__(16) unsigned short sX[WS * SXS];         // X tile, later attn-out
  __shared__ __align__(16) unsigned short sQK[WS * SQS];        // [Q | K] bf16
  __shared__ __align__(16) unsigned short sVt[(NH * HD) * SVS]; // V^T: [h*32+d][tok]
  __shared__ __align__(16) unsigned short sP[8 * 16 * SPS];     // per-wave prob block
  __shared__ float sRel[128];

  const int tid  = threadIdx.x;
  const int wave = __builtin_amdgcn_readfirstlane((int)(threadIdx.x >> 5));  // scalar
  const int lane = tid & 31;
  const int hf   = lane >> 4;
  const int lm   = lane & 15;
  const v8f vzero = {0.f, 0.f, 0.f, 0.f, 0.f, 0.f, 0.f, 0.f};

  // ---------------- Phase 0: stage X window (f32 -> bf16 LDS) ----------------
  const float* xw = x + (size_t)blockIdx.x * (WS * DIM);
#pragma unroll
  for (int it = 0; it < 16; ++it) {
    int idx4 = it * 256 + tid;              // 4096 float4 chunks
    v4f v = ((const v4f*)xw)[idx4];
    int row = idx4 >> 6;
    int col = (idx4 & 63) * 4;
    unsigned* d = (unsigned*)&sX[row * SXS + col];
    d[0] = (unsigned)f2bf(v[0]) | ((unsigned)f2bf(v[1]) << 16);
    d[1] = (unsigned)f2bf(v[2]) | ((unsigned)f2bf(v[3]) << 16);
  }
  if (tid < 2 * WS - 1) sRel[tid] = rel_bias[tid];
  __syncthreads();

  // ---------------- Phase 1: QKV GEMM (64x256 @ 256x768) ----------------
  // Wave w owns output tile columns nt = 6w..6w+5; one B-fragment set (weights)
  // feeds all 4 row tiles. Full unroll -> scheduler prefetches across tiles.
#pragma unroll
  for (int nt6 = 0; nt6 < 6; ++nt6) {
    const int nt = wave * 6 + nt6;          // scalar
    const int ocol = nt * 16 + lm;
    v16bf B[8];
    if (WBF) {
      const unsigned short* wrow = wqkv_bf + (size_t)ocol * DIM;
#pragma unroll
      for (int kc = 0; kc < 8; ++kc) B[kc] = frag_b_bf(wrow, kc * 32, hf);
    } else {
      const float* wrow = w_qkv + (size_t)ocol * DIM;
#pragma unroll
      for (int kc = 0; kc < 8; ++kc) B[kc] = frag_b_w(wrow, kc * 32, hf);
    }
#pragma unroll
    for (int mt = 0; mt < 4; ++mt) {
      const unsigned short* arow = &sX[(mt * 16 + lm) * SXS];
      v8f acc = vzero;
#pragma unroll
      for (int kc = 0; kc < 8; ++kc)
        acc = wmma_bf16(frag_a(arow, kc * 32, hf), B[kc], acc);
      if (nt < 32) {  // Q (nt<16) and K (16<=nt<32): store row-major
#pragma unroll
        for (int i = 0; i < 8; ++i)
          sQK[(mt * 16 + i + hf * 8) * SQS + nt * 16 + lm] = f2bf(acc[i]);
      } else {        // V: store transposed [d][tok]
        int h = (nt - 32) >> 1;
        int dloc = ((nt - 32) & 1) * 16 + lm;
#pragma unroll
        for (int i = 0; i < 8; ++i)
          sVt[(h * HD + dloc) * SVS + mt * 16 + i + hf * 8] = f2bf(acc[i]);
      }
    }
  }
  __syncthreads();

  // ---------------- Phase 2: attention, one head per wave ----------------
  {
    const int h = wave;  // scalar
    v16bf BK[4];
#pragma unroll
    for (int nt = 0; nt < 4; ++nt)   // B[d][k_token]: rows contiguous in d
      BK[nt] = frag_b_bf(&sQK[(nt * 16 + lm) * SQS + DIM + h * HD], 0, hf);
    v16bf BV[2][2];
#pragma unroll
    for (int kc = 0; kc < 2; ++kc)
#pragma unroll
      for (int dt = 0; dt < 2; ++dt)  // B[k][d] from V^T rows (contiguous k)
        BV[kc][dt] = frag_b_bf(&sVt[(h * HD + dt * 16 + lm) * SVS], kc * 32, hf);
    unsigned short* sPw = &sP[wave * 16 * SPS];

#pragma unroll
    for (int mt = 0; mt < 4; ++mt) {
      v16bf AQ = frag_a(&sQK[(mt * 16 + lm) * SQS + h * HD], 0, hf);
      v8f S[4];
#pragma unroll
      for (int nt = 0; nt < 4; ++nt) S[nt] = wmma_bf16(AQ, BK[nt], vzero);
      // row-block softmax (rows mt*16 .. mt*16+15 complete across 4 nt tiles)
#pragma unroll
      for (int i = 0; i < 8; ++i) {
        int q = mt * 16 + i + hf * 8;
        float t[4];
        float mx = -3.0e38f;
#pragma unroll
        for (int nt = 0; nt < 4; ++nt) {
          int k = nt * 16 + lm;
          float s = S[nt][i] * QKSCALE + sRel[k - q + WS - 1];
          t[nt] = s;
          mx = fmaxf(mx, s);
        }
#pragma unroll
        for (int m = 1; m < 16; m <<= 1) mx = fmaxf(mx, __shfl_xor(mx, m, 32));
        float sum = 0.f;
#pragma unroll
        for (int nt = 0; nt < 4; ++nt) { t[nt] = __expf(t[nt] - mx); sum += t[nt]; }
#pragma unroll
        for (int m = 1; m < 16; m <<= 1) sum += __shfl_xor(sum, m, 32);
        float inv = 1.0f / sum;
#pragma unroll
        for (int nt = 0; nt < 4; ++nt)
          sPw[(i + hf * 8) * SPS + nt * 16 + lm] = f2bf(t[nt] * inv);
      }
      // AV: out(16x32) = P(16x64) @ V(64x32)
      v8f O0 = vzero, O1 = vzero;
#pragma unroll
      for (int kc = 0; kc < 2; ++kc) {
        v16bf AP = frag_a(&sPw[lm * SPS], kc * 32, hf);
        O0 = wmma_bf16(AP, BV[kc][0], O0);
        O1 = wmma_bf16(AP, BV[kc][1], O1);
      }
#pragma unroll
      for (int i = 0; i < 8; ++i) {
        int r = (mt * 16 + i + hf * 8) * SXS + h * HD + lm;
        sX[r]      = f2bf(O0[i]);
        sX[r + 16] = f2bf(O1[i]);
      }
    }
  }
  __syncthreads();

  // ---------------- Phase 3: out-proj (64x256 @ 256x256) + bias ----------------
  float* outw = out + (size_t)blockIdx.x * (WS * DIM);
#pragma unroll
  for (int nt2 = 0; nt2 < 2; ++nt2) {
    const int nt = wave * 2 + nt2;          // scalar
    const int ocol = nt * 16 + lm;
    float bias = b_proj[ocol];
    v16bf B[8];
    if (WBF) {
      const unsigned short* wrow = wproj_bf + (size_t)ocol * DIM;
#pragma unroll
      for (int kc = 0; kc < 8; ++kc) B[kc] = frag_b_bf(wrow, kc * 32, hf);
    } else {
      const float* wrow = w_proj + (size_t)ocol * DIM;
#pragma unroll
      for (int kc = 0; kc < 8; ++kc) B[kc] = frag_b_w(wrow, kc * 32, hf);
    }
#pragma unroll
    for (int mt = 0; mt < 4; ++mt) {
      const unsigned short* arow = &sX[(mt * 16 + lm) * SXS];
      v8f acc = vzero;
#pragma unroll
      for (int kc = 0; kc < 8; ++kc)
        acc = wmma_bf16(frag_a(arow, kc * 32, hf), B[kc], acc);
#pragma unroll
      for (int i = 0; i < 8; ++i)
        outw[(mt * 16 + i + hf * 8) * DIM + ocol] = acc[i] + bias;
    }
  }
}

extern "C" void kernel_launch(void* const* d_in, const int* in_sizes, int n_in,
                              void* d_out, int out_size, void* d_ws, size_t ws_size,
                              hipStream_t stream) {
  const float* x        = (const float*)d_in[0];
  const float* w_qkv    = (const float*)d_in[1];
  const float* w_proj   = (const float*)d_in[2];
  const float* b_proj   = (const float*)d_in[3];
  const float* rel_bias = (const float*)d_in[4];
  float* outp = (float*)d_out;
  (void)n_in; (void)out_size;

  int nwin = in_sizes[0] / (WS * DIM);  // 2048 windows (L % WS == 0)

  if (ws_size >= WS_BYTES_NEEDED) {
    unsigned short* wbf = (unsigned short*)d_ws;
    // 65536 threads x 4 elements = 262144 weight elements
    conv_weights_kernel<<<256, 256, 0, stream>>>(w_qkv, w_proj, wbf);
    lwa_fused_kernel<true><<<nwin, 256, 0, stream>>>(
        x, w_qkv, w_proj, b_proj, rel_bias, wbf, wbf + WQKV_ELEMS, outp);
  } else {
    lwa_fused_kernel<false><<<nwin, 256, 0, stream>>>(
        x, w_qkv, w_proj, b_proj, rel_bias, nullptr, nullptr, outp);
  }
}
